// MultiHotVQVAEQuantizer_9998683865096
// MI455X (gfx1250) — compile-verified, matrix operand order checked by reading, not run on hardware
//
#include <hip/hip_runtime.h>

// ---------------------------------------------------------------------------
// MultiHot VQ-VAE quantizer for gfx1250 (MI455X), wave32 + WMMA bf16.
// Outputs (concatenated in d_out): z_q_ste [N*D] f32, loss [1] f32, k_hot [N*Q] f32.
//
// Score GEMM uses split-precision bf16x3 (hi*hi + hi*lo + lo*hi) on
// v_wmma_f32_16x16x32_bf16. The embedding's hi/lo split is precomputed once
// into workspace in B-fragment lane layout so the hot loop is loads + WMMA.
// ---------------------------------------------------------------------------

typedef __attribute__((ext_vector_type(16))) __bf16 v16bf;
typedef __attribute__((ext_vector_type(8)))  float  v8f;
typedef __attribute__((ext_vector_type(4)))  float  f4;

#define N_TOK 16384
#define DIM   256
#define QSZ   8192
#define KTOP  15
#define CCOST 0.25f

#define NWAVE 8          // 256 threads / 32
#define MTILE 16         // tokens per workgroup
#define QTILES (QSZ / 16)
#define KCHUNKS (DIM / 32)

__device__ __forceinline__ float pinf() { return __builtin_inff(); }

// ---------------------------------------------------------------------------
// Kernel 1: e_norm[q] = sum_d emb[q][d]^2
// ---------------------------------------------------------------------------
__global__ __launch_bounds__(256) void vq_enorm(const float* __restrict__ emb,
                                                float* __restrict__ e_norm) {
  int q = blockIdx.x * 256 + threadIdx.x;
  if (q >= QSZ) return;
  const f4* r = (const f4*)(emb + (size_t)q * DIM);
  float s = 0.f;
  #pragma unroll 8
  for (int i = 0; i < DIM / 4; ++i) {
    f4 v = r[i];
    s += v.x * v.x + v.y * v.y + v.z * v.z + v.w * v.w;
  }
  e_norm[q] = s;
}

// ---------------------------------------------------------------------------
// Kernel 1b: pre-split embedding into bf16 hi/lo B-fragments.
// Record per (tile t, chunk c, lane l): 16 hi halfs + 16 lo halfs (64 B),
// half h -> K = (h&7) + 16*(h>>3) + 8*(l>>4), N(col q) = t*16 + (l&15).
// ---------------------------------------------------------------------------
__global__ __launch_bounds__(256) void vq_bfrag(const float* __restrict__ emb,
                                                __bf16* __restrict__ ebf) {
  int tid = blockIdx.x * 256 + threadIdx.x;      // over QTILES*KCHUNKS*32
  if (tid >= QTILES * KCHUNKS * 32) return;
  const int lane = tid & 31;
  const int c    = (tid >> 5) & (KCHUNKS - 1);
  const int t    = tid >> 8;
  const int q    = t * 16 + (lane & 15);
  const int kb   = c * 32 + ((lane >> 4) << 3);

  const float* src = emb + (size_t)q * DIM + kb;
  f4 f0 = *(const f4*)(src);
  f4 f1 = *(const f4*)(src + 4);
  f4 f2 = *(const f4*)(src + 16);
  f4 f3 = *(const f4*)(src + 20);
  float fv[16];
  *(f4*)(fv + 0) = f0; *(f4*)(fv + 4) = f1;
  *(f4*)(fv + 8) = f2; *(f4*)(fv + 12) = f3;

  v16bf hi, lo;
  #pragma unroll
  for (int h = 0; h < 16; ++h) {
    __bf16 bh = (__bf16)fv[h];
    hi[h] = bh;
    lo[h] = (__bf16)(fv[h] - (float)bh);
  }
  __bf16* dst = ebf + (size_t)tid * 32;          // tid == ((t*KCHUNKS+c)*32+lane)
  *(v16bf*)(dst)      = hi;
  *(v16bf*)(dst + 16) = lo;
}

// ---------------------------------------------------------------------------
// Kernel 2: fused score-GEMM + per-row top-K + outputs.
// One workgroup (8 waves) per 16 tokens. PRE: use pre-split B fragments.
// ---------------------------------------------------------------------------
template <bool PRE>
__global__ __launch_bounds__(256, 1) void vq_main(
    const float* __restrict__ z_e, const float* __restrict__ emb,
    const __bf16* __restrict__ ebf, const float* __restrict__ e_norm,
    float* __restrict__ out_zq, float* __restrict__ out_khot,
    float* __restrict__ ws_part) {
  __shared__ float zblk[MTILE * DIM];           // 16 KB staged z block (fp32)
  __shared__ float sc[NWAVE * 16 * 17];         // score tiles, pad 17 vs banks
  __shared__ float topv[NWAVE * 16 * 16];       // per-(wave,row) top-16 values
  __shared__ int   topi[NWAVE * 16 * 16];       // per-(wave,row) top-16 indices
  __shared__ int   fidx[16 * KTOP];             // final top-15 per row
  __shared__ float red[256];

  const int tid    = threadIdx.x;
  const int wave   = tid >> 5;
  const int lane   = tid & 31;
  const int laneHi = lane >> 4;                 // K-half select (ISA layout)
  const int nlo    = lane & 15;                 // M (A) / N (B) index
  const int tok0   = blockIdx.x * MTILE;

  // ---- stage z block to LDS; init top-K lists ----
  {
    const f4* src = (const f4*)(z_e + (size_t)tok0 * DIM);
    for (int i = tid; i < MTILE * DIM / 4; i += 256) ((f4*)zblk)[i] = src[i];
  }
  for (int i = tid; i < NWAVE * 16 * 16; i += 256) {
    topv[i] = pinf();
    topi[i] = 0;
  }
  __syncthreads();

  // ---- build A fragments (16x32 bf16, hi/lo split) once per wave ----
  // A layout: lanes 0-15 hold M=0-15; half h -> K = (h&7) + 16*(h>>3) + 8*laneHi
  v16bf a_hi[KCHUNKS], a_lo[KCHUNKS];
  #pragma unroll
  for (int c = 0; c < KCHUNKS; ++c) {
    const int kb = c * 32 + (laneHi << 3);
    #pragma unroll
    for (int h = 0; h < 16; ++h) {
      int k = kb + (h & 7) + ((h >> 3) << 4);
      float f = zblk[nlo * DIM + k];
      __bf16 hi = (__bf16)f;
      a_hi[c][h] = hi;
      a_lo[c][h] = (__bf16)(f - (float)hi);
    }
  }

  float tmax = pinf();   // threshold = worst kept value (lanes 0-15 only)
  int   tpos = 0;
  const int lbase = (wave * 16 + nlo) * 16;

  // ---- main loop: wave-strided Q tiles ----
  for (int t = wave; t < QTILES; t += NWAVE) {
    const int qb = t * 16;
    v8f acc = {0.f, 0.f, 0.f, 0.f, 0.f, 0.f, 0.f, 0.f};
    const float* brow = emb + (size_t)(qb + nlo) * DIM;
    const v16bf* bfr =
        (const v16bf*)(ebf) + ((size_t)t * KCHUNKS * 32 + lane) * 2;

    #pragma unroll
    for (int c = 0; c < KCHUNKS; ++c) {
      v16bf b_hi, b_lo;
      if (PRE) {
        // pre-split fragments: pure b128 loads, no VALU
        b_hi = bfr[(size_t)c * 64 + 0];
        b_lo = bfr[(size_t)c * 64 + 1];
      } else {
        const int kb = c * 32 + (laneHi << 3);
        f4 f0 = *(const f4*)(brow + kb);
        f4 f1 = *(const f4*)(brow + kb + 4);
        f4 f2 = *(const f4*)(brow + kb + 16);
        f4 f3 = *(const f4*)(brow + kb + 20);
        float fv[16];
        *(f4*)(fv + 0) = f0; *(f4*)(fv + 4) = f1;
        *(f4*)(fv + 8) = f2; *(f4*)(fv + 12) = f3;
        #pragma unroll
        for (int h = 0; h < 16; ++h) {
          __bf16 hi = (__bf16)fv[h];
          b_hi[h] = hi;
          b_lo[h] = (__bf16)(fv[h] - (float)hi);
        }
      }
      // split-precision fp32 emulation: hi*hi + hi*lo + lo*hi
      acc = __builtin_amdgcn_wmma_f32_16x16x32_bf16(false, a_hi[c], false, b_hi,
                                                    (short)0, acc, false, false);
      acc = __builtin_amdgcn_wmma_f32_16x16x32_bf16(false, a_hi[c], false, b_lo,
                                                    (short)0, acc, false, false);
      acc = __builtin_amdgcn_wmma_f32_16x16x32_bf16(false, a_lo[c], false, b_hi,
                                                    (short)0, acc, false, false);
    }

    // score = ||e||^2 - 2*dot ; C layout: VGPR r, lanes 0-15 -> M=r, N=lane;
    // lanes 16-31 -> M=r+8, N=lane-16
    float en = e_norm[qb + nlo];
    #pragma unroll
    for (int r = 0; r < 8; ++r) {
      int m = r + (laneHi << 3);
      sc[(wave * 16 + m) * 17 + nlo] = en - 2.0f * acc[r];
    }
    // per-row top-K insertion scan: lanes 0-15, row = lane (wave-synchronous,
    // same-wave LDS RAW is in-order)
    if (lane < 16) {
      #pragma unroll
      for (int n = 0; n < 16; ++n) {
        float cand = sc[(wave * 16 + lane) * 17 + n];
        if (cand < tmax) {
          topv[lbase + tpos] = cand;
          topi[lbase + tpos] = qb + n;
          tmax = topv[lbase]; tpos = 0;
          #pragma unroll
          for (int j = 1; j < 16; ++j) {
            float v = topv[lbase + j];
            if (v > tmax) { tmax = v; tpos = j; }
          }
        }
      }
    }
  }
  __syncthreads();

  // ---- merge 8 per-wave lists -> final top-15 per row ----
  if (tid < 16) {
    const int row = tid;
    for (int k = 0; k < KTOP; ++k) {
      float best = pinf();
      int bpos = row * 16;
      for (int w = 0; w < NWAVE; ++w) {
        int base = (w * 16 + row) * 16;
        #pragma unroll
        for (int j = 0; j < 16; ++j) {
          float v = topv[base + j];
          if (v < best) { best = v; bpos = base + j; }
        }
      }
      fidx[row * KTOP + k] = topi[bpos];
      topv[bpos] = pinf();
    }
  }
  __syncthreads();

  // ---- k_hot: zero-fill then scatter ones (slab is only 4B-aligned) ----
  float* khrow = out_khot + (size_t)tok0 * QSZ;
  for (int i = tid; i < MTILE * QSZ; i += 256) khrow[i] = 0.0f;
  __syncthreads();
  if (tid < 16 * KTOP) {
    int row = tid / KTOP, k = tid % KTOP;
    khrow[(size_t)row * QSZ + fidx[row * KTOP + k]] = 1.0f;
  }

  // ---- z_q_ste = sum of top-15 embeddings; loss partial ----
  const int row = tid >> 4;           // 16 threads per row
  const int d0  = (tid & 15) * 16;    // 16 dims per thread
  float zq[16];
  #pragma unroll
  for (int j = 0; j < 16; ++j) zq[j] = 0.f;
  for (int k = 0; k < KTOP; ++k) {
    const float* er = emb + (size_t)fidx[row * KTOP + k] * DIM + d0;
    #pragma unroll
    for (int j = 0; j < 16; j += 4) {
      f4 v = *(const f4*)(er + j);
      zq[j] += v.x; zq[j + 1] += v.y; zq[j + 2] += v.z; zq[j + 3] += v.w;
    }
  }
  float lpart = 0.f;
  #pragma unroll
  for (int j = 0; j < 16; ++j) {
    float dz = zq[j] - zblk[row * DIM + d0 + j];
    lpart += dz * dz;
  }
  float* zo = out_zq + (size_t)(tok0 + row) * DIM + d0;
  #pragma unroll
  for (int j = 0; j < 16; j += 4) {
    f4 v; v.x = zq[j]; v.y = zq[j + 1]; v.z = zq[j + 2]; v.w = zq[j + 3];
    *(f4*)(zo + j) = v;
  }

  // ---- deterministic block reduction of loss partial ----
  red[tid] = lpart;
  __syncthreads();
  for (int s = 128; s > 0; s >>= 1) {
    if (tid < s) red[tid] += red[tid + s];
    __syncthreads();
  }
  if (tid == 0) ws_part[blockIdx.x] = red[0];
}

// ---------------------------------------------------------------------------
// Kernel 3: deterministic final loss reduction
// ---------------------------------------------------------------------------
__global__ __launch_bounds__(256) void vq_loss_final(const float* __restrict__ part,
                                                     float* __restrict__ out_loss) {
  __shared__ float red[256];
  float s = 0.f;
  for (int i = threadIdx.x; i < N_TOK / MTILE; i += 256) s += part[i];
  red[threadIdx.x] = s;
  __syncthreads();
  for (int st = 128; st > 0; st >>= 1) {
    if (threadIdx.x < st) red[threadIdx.x] += red[threadIdx.x + st];
    __syncthreads();
  }
  if (threadIdx.x == 0)
    *out_loss = (1.0f + CCOST) * red[0] / (float)((size_t)N_TOK * DIM);
}

// ---------------------------------------------------------------------------
extern "C" void kernel_launch(void* const* d_in, const int* in_sizes, int n_in,
                              void* d_out, int out_size, void* d_ws, size_t ws_size,
                              hipStream_t stream) {
  (void)in_sizes; (void)n_in; (void)out_size;
  const float* z_e = (const float*)d_in[0];
  const float* emb = (const float*)d_in[1];

  float* out      = (float*)d_out;
  float* out_zq   = out;                              // [N*D]
  float* out_loss = out + (size_t)N_TOK * DIM;        // [1]
  float* out_khot = out + (size_t)N_TOK * DIM + 1;    // [N*Q]

  // ws layout: e_norm [Q] | parts [N/16] | (64B-aligned) ebf hi/lo fragments
  float*  e_norm  = (float*)d_ws;
  float*  parts   = e_norm + QSZ;
  size_t  ebf_off = ((size_t)(QSZ + N_TOK / MTILE) * sizeof(float) + 63) & ~(size_t)63;
  __bf16* ebf     = (__bf16*)((char*)d_ws + ebf_off);
  size_t  need    = ebf_off + (size_t)QSZ * DIM * 2 * sizeof(__bf16);
  const bool pre  = ws_size >= need;

  vq_enorm<<<QSZ / 256, 256, 0, stream>>>(emb, e_norm);
  if (pre) {
    vq_bfrag<<<(QTILES * KCHUNKS * 32) / 256, 256, 0, stream>>>(emb, ebf);
    vq_main<true><<<N_TOK / MTILE, 256, 0, stream>>>(z_e, emb, ebf, e_norm,
                                                     out_zq, out_khot, parts);
  } else {
    vq_main<false><<<N_TOK / MTILE, 256, 0, stream>>>(z_e, emb, ebf, e_norm,
                                                      out_zq, out_khot, parts);
  }
  vq_loss_final<<<1, 256, 0, stream>>>(parts, out_loss);
}
